// ExtSummModel_60584808677677
// MI455X (gfx1250) — compile-verified, matrix operand order checked by reading, not run on hardware
//
#include <hip/hip_runtime.h>
#include <hip/hip_bf16.h>
#include <math.h>

// ---------------------------------------------------------------------------
// ExtSumm forward for MI455X (gfx1250, wave32, WMMA).
// GEMMs use v_wmma_f32_16x16x32_bf16 (bf16 A/B, f32 accumulate).
// ---------------------------------------------------------------------------

#define BB 32
#define SS 256
#define LL 24
#define EE 300
#define EP 320      // E padded to multiple of 32 for WMMA K
#define HH 256
#define GG 768      // 3*H
#define TT 16
#define DD 256
#define FH 1024     // 4*H

typedef __attribute__((ext_vector_type(16))) __bf16 v16bf;
typedef __attribute__((ext_vector_type(8)))  float  v8f;

// ---- WMMA fragment helpers (CDNA5 16x16x32 bf16 layout) -------------------
// A (MxK): lane l -> row = l%16 ; halfs[0..7] = K kb..kb+7, halfs[8..15] = K kb+16..kb+23,
//          kb = 8*(l/16).  B (KxN) loaded from weight stored [N][Kp]: identical pattern
//          with N playing the role of M (K contiguous in memory).
__device__ __forceinline__ v16bf load_frag(const __bf16* p0, int ld, int r0, int k0, int lane) {
  const __bf16* p = p0 + (size_t)(r0 + (lane & 15)) * ld + k0 + ((lane >> 4) << 3);
  v16bf f;
#pragma unroll
  for (int i = 0; i < 8; ++i) { f[i] = p[i]; f[i + 8] = p[i + 16]; }
  return f;
}

__device__ __forceinline__ v8f wmma_bf16(v16bf a, v16bf b, v8f c) {
  return __builtin_amdgcn_wmma_f32_16x16x32_bf16(false, a, false, b, (short)0, c, false, false);
}

// ---- weight conversion ----------------------------------------------------
// src [N][K] f32 (K contiguous) -> dst [N][Kp] bf16, zero-padded K
__global__ void k_conv_nk(const float* __restrict__ src, __bf16* __restrict__ dst,
                          int N, int K, int Kp) {
  int idx = blockIdx.x * blockDim.x + threadIdx.x;
  if (idx >= N * Kp) return;
  int n = idx / Kp, k = idx % Kp;
  dst[idx] = (k < K) ? (__bf16)src[(size_t)n * K + k] : (__bf16)0.f;
}

// src [K][N] f32 -> dst [N][Kp] bf16 (transpose), zero-padded K
__global__ void k_conv_kn_T(const float* __restrict__ src, __bf16* __restrict__ dst,
                            int N, int K, int Kp) {
  int idx = blockIdx.x * blockDim.x + threadIdx.x;
  if (idx >= N * Kp) return;
  int n = idx / Kp, k = idx % Kp;
  dst[idx] = (k < K) ? (__bf16)src[(size_t)k * N + n] : (__bf16)0.f;
}

// ---- stage 1: embedding gather + mean -> bf16 [B*S][EP] -------------------
__global__ void k_embed_mean(const int* __restrict__ word_ids, const float* __restrict__ emb,
                             __bf16* __restrict__ sentA) {
  int m = blockIdx.x;            // b*S + s
  int c = threadIdx.x;           // 0..EP-1 (320 threads = 10 waves)
  float acc = 0.f;
  if (c < EE) {
    const int* w = word_ids + (size_t)m * LL;
    for (int l = 0; l < LL; ++l) acc += emb[(size_t)w[l] * EE + c];
    acc *= (1.f / (float)LL);
  }
  sentA[(size_t)m * EP + c] = (__bf16)acc;
}

// ---- generic WMMA GEMM: C[M][N] = A[M][Kp] * Bw[N][Kp]^T + bias -----------
__global__ void k_gemm_bias(const __bf16* __restrict__ A, const __bf16* __restrict__ Bw,
                            const float* __restrict__ bias, float* __restrict__ C,
                            int N, int Kp) {
  int m0 = blockIdx.x * 16, n0 = blockIdx.y * 16;
  int lane = threadIdx.x & 31;
  v8f acc = {};
  for (int k0 = 0; k0 < Kp; k0 += 32) {
    v16bf a = load_frag(A, Kp, m0, k0, lane);
    v16bf b = load_frag(Bw, Kp, n0, k0, lane);
    acc = wmma_bf16(a, b, acc);
  }
  int col = n0 + (lane & 15);
  int rbase = (lane >> 4) << 3;
  float bv = bias ? bias[col] : 0.f;
#pragma unroll
  for (int r = 0; r < 8; ++r)
    C[(size_t)(m0 + rbase + r) * N + col] = acc[r] + bv;
}

// ---- stage 3: one GRU recurrence step (both dirs, 2 M-tiles) --------------
// grid (2 mtiles, 2 dirs), 512 threads = 16 waves. gh = h @ Whh^T via WMMA in
// LDS, then fused gate elementwise update. gi (with bih) precomputed.
__global__ void k_gru_step(int t,
                           const float* __restrict__ giF, const float* __restrict__ giB,
                           float* __restrict__ hF, float* __restrict__ hB,
                           const __bf16* __restrict__ WhhFb, const __bf16* __restrict__ WhhBb,
                           const float* __restrict__ bhhF, const float* __restrict__ bhhB,
                           float* __restrict__ sent_rep) {
  int mtile = blockIdx.x, dir = blockIdx.y;
  const float*  gi  = dir ? giB  : giF;
  float*        h   = dir ? hB   : hF;
  const __bf16* Whh = dir ? WhhBb : WhhFb;
  const float*  bhh = dir ? bhhB : bhhF;
  int s = dir ? (SS - 1 - t) : t;

  __shared__ __bf16 hs[16 * HH];    // 8 KB: h tile in bf16
  __shared__ float  ghs[16 * GG];   // 48 KB: gh tile
  int m0 = mtile * 16;
  int tid = threadIdx.x;

  for (int idx = tid; idx < 16 * HH; idx += blockDim.x) {
    int r = idx >> 8, c = idx & 255;
    hs[idx] = (__bf16)h[(m0 + r) * HH + c];
  }
  __syncthreads();

  int wave = tid >> 5, lane = tid & 31;
#pragma unroll
  for (int j = 0; j < 3; ++j) {               // 16 waves x 3 tiles = 48 N-tiles
    int n0 = (wave * 3 + j) * 16;
    v8f acc = {};
    for (int k0 = 0; k0 < HH; k0 += 32) {
      v16bf a = load_frag(hs, HH, 0, k0, lane);
      v16bf b = load_frag(Whh, HH, n0, k0, lane);
      acc = wmma_bf16(a, b, acc);
    }
    int col = n0 + (lane & 15);
    int rbase = (lane >> 4) << 3;
#pragma unroll
    for (int r = 0; r < 8; ++r) ghs[(rbase + r) * GG + col] = acc[r];
  }
  __syncthreads();

  for (int idx = tid; idx < 16 * HH; idx += blockDim.x) {
    int r = idx >> 8, c = idx & 255;
    int gm = m0 + r;
    const float* gir = gi + ((size_t)gm * SS + s) * GG;
    float ir = gir[c], iz = gir[HH + c], in = gir[2 * HH + c];
    float hr = ghs[r * GG + c] + bhh[c];
    float hz = ghs[r * GG + HH + c] + bhh[HH + c];
    float hn = ghs[r * GG + 2 * HH + c] + bhh[2 * HH + c];
    float rg = 1.f / (1.f + expf(-(ir + hr)));
    float zg = 1.f / (1.f + expf(-(iz + hz)));
    float ng = tanhf(in + rg * hn);
    float h2 = (1.f - zg) * ng + zg * h[gm * HH + c];
    h[gm * HH + c] = h2;
    sent_rep[((size_t)gm * SS + s) * 512 + dir * HH + c] = h2;
  }
}

// ---- doc_vec: faithful torch reshape of hidden [2,B,H] -> [B,2H] ----------
__global__ void k_doc_vec(const float* __restrict__ hF, const float* __restrict__ hB,
                          float* __restrict__ doc_vec) {
  int idx = blockIdx.x * blockDim.x + threadIdx.x;   // 32*512
  if (idx >= BB * 2 * HH) return;
  int dir = idx >> 13;           // flat / (B*H)
  int rem = idx & 8191;
  int bb = rem >> 8, hh = rem & 255;
  doc_vec[idx] = (dir ? hB : hF)[bb * HH + hh];
}

// ---- topic_mat via boundary differencing on padded sent_rep ---------------
__global__ void k_topic_mat(const float* __restrict__ sent_rep, const int* __restrict__ tse,
                            float* __restrict__ topic_mat) {
  int b = blockIdx.x;
  for (int idx = threadIdx.x; idx < TT * 512; idx += blockDim.x) {
    int t = idx >> 9, c = idx & 511;
    int st = tse[(b * TT + t) * 2 + 0];
    int en = tse[(b * TT + t) * 2 + 1];
    // padded[i]: i in 1..S -> sent_rep[i-1]; i==0 or i==S+1 -> 0
    auto pad = [&](int i) -> float {
      if (i <= 0 || i > SS) return 0.f;
      return sent_rep[((size_t)b * SS + (i - 1)) * 512 + c];
    };
    float val = (c < HH) ? (pad(en) - pad(st - 1)) : (pad(st) - pad(en + 1));
    topic_mat[((size_t)b * TT + t) * 512 + c] = val;
  }
}

// ---- build cat_ds / cat_ts (bf16) and topic_rep (f32) ---------------------
__global__ void k_cat(const float* __restrict__ sent_rep, const float* __restrict__ doc_vec,
                      const float* __restrict__ topic_mat, const int* __restrict__ tse,
                      __bf16* __restrict__ catDS, __bf16* __restrict__ catTS,
                      float* __restrict__ topic_rep) {
  int m = blockIdx.x;            // b*S + s
  int b = m / SS, s = m % SS;
  // searchsorted(ends, s+1, left): first t with ends[t] >= s+1
  int tt = TT - 1;
  for (int t = 0; t < TT; ++t)
    if (tse[(b * TT + t) * 2 + 1] >= s + 1) { tt = t; break; }
  for (int j = threadIdx.x; j < 512; j += blockDim.x) {
    float sr = sent_rep[(size_t)m * 512 + j];
    float dv = doc_vec[b * 512 + j];
    float tr = topic_mat[((size_t)b * TT + tt) * 512 + j];
    catDS[(size_t)m * FH + j]       = (__bf16)dv;
    catDS[(size_t)m * FH + 512 + j] = (__bf16)sr;
    catTS[(size_t)m * FH + j]       = (__bf16)tr;
    catTS[(size_t)m * FH + 512 + j] = (__bf16)sr;
    topic_rep[(size_t)m * 512 + j]  = tr;
  }
}

// ---- fused attention score: tanh(X @ W_att) @ v_att (no materialization) --
// grid (512 Mtiles, 2 matrices), 256 threads = 8 waves x 8 N-tiles, K=1024.
__global__ void k_att_score(const __bf16* __restrict__ catDS, const __bf16* __restrict__ catTS,
                            const __bf16* __restrict__ WattT, const float* __restrict__ v_att,
                            float* __restrict__ scoresD, float* __restrict__ scoresT) {
  int mt = blockIdx.x, which = blockIdx.y;
  const __bf16* X = which ? catTS : catDS;
  float* out = which ? scoresT : scoresD;
  __shared__ float red[16];
  int tid = threadIdx.x, wave = tid >> 5, lane = tid & 31;
  if (tid < 16) red[tid] = 0.f;
  __syncthreads();
  int m0 = mt * 16;
  float part[8];
#pragma unroll
  for (int r = 0; r < 8; ++r) part[r] = 0.f;
  for (int j = 0; j < 8; ++j) {
    int n0 = (wave * 8 + j) * 16;
    v8f acc = {};
    for (int k0 = 0; k0 < FH; k0 += 32) {
      v16bf a = load_frag(X, FH, m0, k0, lane);
      v16bf b = load_frag(WattT, FH, n0, k0, lane);
      acc = wmma_bf16(a, b, acc);
    }
    float v = v_att[n0 + (lane & 15)];
#pragma unroll
    for (int r = 0; r < 8; ++r) part[r] += tanhf(acc[r]) * v;
  }
  int rbase = (lane >> 4) << 3;
#pragma unroll
  for (int r = 0; r < 8; ++r) atomicAdd(&red[rbase + r], part[r]);
  __syncthreads();
  if (tid < 16) out[m0 + tid] = red[tid];
}

// ---- softmax over S per batch row -----------------------------------------
__global__ void k_softmax(const float* __restrict__ scoresD, const float* __restrict__ scoresT,
                          float* __restrict__ wD, float* __restrict__ wT) {
  int b = blockIdx.x, which = blockIdx.y, tid = threadIdx.x;  // blockDim = S = 256
  const float* s = (which ? scoresT : scoresD) + (size_t)b * SS;
  float* w = (which ? wT : wD) + (size_t)b * SS;
  __shared__ float sh[SS];
  float v = s[tid];
  sh[tid] = v; __syncthreads();
  for (int st = SS / 2; st > 0; st >>= 1) {
    if (tid < st) sh[tid] = fmaxf(sh[tid], sh[tid + st]);
    __syncthreads();
  }
  float mx = sh[0]; __syncthreads();
  float e = expf(v - mx);
  sh[tid] = e; __syncthreads();
  for (int st = SS / 2; st > 0; st >>= 1) {
    if (tid < st) sh[tid] += sh[tid + st];
    __syncthreads();
  }
  w[tid] = e / sh[0];
}

// ---- context + decoder input (bf16) ---------------------------------------
__global__ void k_context_inp(const float* __restrict__ sent_rep, const float* __restrict__ doc_vec,
                              const float* __restrict__ topic_rep, const float* __restrict__ wD,
                              const float* __restrict__ wT, __bf16* __restrict__ inp) {
  int m = blockIdx.x;
  int b = m / SS;
  float dw = wD[m], tw = wT[m];
  for (int j = threadIdx.x; j < 512; j += blockDim.x) {
    float ctx = dw * doc_vec[b * 512 + j] + tw * topic_rep[(size_t)m * 512 + j];
    inp[(size_t)m * FH + j]       = (__bf16)sent_rep[(size_t)m * 512 + j];
    inp[(size_t)m * FH + 512 + j] = (__bf16)ctx;
  }
}

// ---- fused dense head: relu(inp @ W_dna + b) @ W_out + b_out -> logits ----
// grid 512 Mtiles, 256 threads = 8 waves x 2 N-tiles (N=256), K=1024.
__global__ void k_dense_out(const __bf16* __restrict__ inp, const __bf16* __restrict__ WdnaT,
                            const float* __restrict__ b_dna, const float* __restrict__ W_out,
                            const float* __restrict__ b_out, float* __restrict__ logits) {
  int mt = blockIdx.x;
  __shared__ float red[16];
  int tid = threadIdx.x, wave = tid >> 5, lane = tid & 31;
  if (tid < 16) red[tid] = 0.f;
  __syncthreads();
  int m0 = mt * 16;
  float part[8];
#pragma unroll
  for (int r = 0; r < 8; ++r) part[r] = 0.f;
  for (int j = 0; j < 2; ++j) {
    int n0 = (wave * 2 + j) * 16;
    v8f acc = {};
    for (int k0 = 0; k0 < FH; k0 += 32) {
      v16bf a = load_frag(inp, FH, m0, k0, lane);
      v16bf b = load_frag(WdnaT, FH, n0, k0, lane);
      acc = wmma_bf16(a, b, acc);
    }
    int col = n0 + (lane & 15);
    float bb = b_dna[col], wo = W_out[col];
#pragma unroll
    for (int r = 0; r < 8; ++r) {
      float h = acc[r] + bb;
      part[r] += (h > 0.f ? h : 0.f) * wo;
    }
  }
  int rbase = (lane >> 4) << 3;
#pragma unroll
  for (int r = 0; r < 8; ++r) atomicAdd(&red[rbase + r], part[r]);
  __syncthreads();
  if (tid < 16) logits[m0 + tid] = red[tid] + b_out[0];
}

// ===========================================================================
extern "C" void kernel_launch(void* const* d_in, const int* in_sizes, int n_in,
                              void* d_out, int out_size, void* d_ws, size_t ws_size,
                              hipStream_t stream) {
  (void)in_sizes; (void)n_in; (void)out_size; (void)ws_size;
  const int*   word_ids = (const int*)d_in[0];
  const int*   tse      = (const int*)d_in[1];
  const float* emb      = (const float*)d_in[2];
  const float* Wih_f    = (const float*)d_in[3];
  const float* Whh_f    = (const float*)d_in[4];
  const float* bih_f    = (const float*)d_in[5];
  const float* bhh_f    = (const float*)d_in[6];
  const float* Wih_b    = (const float*)d_in[7];
  const float* Whh_b    = (const float*)d_in[8];
  const float* bih_b    = (const float*)d_in[9];
  const float* bhh_b    = (const float*)d_in[10];
  const float* W_att    = (const float*)d_in[11];
  const float* v_att    = (const float*)d_in[12];
  const float* W_dna    = (const float*)d_in[13];
  const float* b_dna    = (const float*)d_in[14];
  const float* W_out    = (const float*)d_in[15];
  const float* b_out    = (const float*)d_in[16];

  const int M = BB * SS;  // 8192

  // ---- workspace bump allocator (256B aligned) ----------------------------
  char* ws = (char*)d_ws;
  size_t off = 0;
  auto take = [&](size_t bytes) -> void* {
    void* p = ws + off;
    off += (bytes + 255) & ~(size_t)255;
    return p;
  };
  __bf16* sentA  = (__bf16*)take((size_t)M * EP * 2);
  __bf16* WihFb  = (__bf16*)take((size_t)GG * EP * 2);
  __bf16* WihBb  = (__bf16*)take((size_t)GG * EP * 2);
  __bf16* WhhFb  = (__bf16*)take((size_t)GG * HH * 2);
  __bf16* WhhBb  = (__bf16*)take((size_t)GG * HH * 2);
  __bf16* WattTb = (__bf16*)take((size_t)FH * FH * 2);
  __bf16* WdnaTb = (__bf16*)take((size_t)DD * FH * 2);
  float* giF      = (float*)take((size_t)M * GG * 4);
  float* giB      = (float*)take((size_t)M * GG * 4);
  float* sent_rep = (float*)take((size_t)M * 512 * 4);
  float* hF       = (float*)take((size_t)BB * HH * 4);
  float* hB       = (float*)take((size_t)BB * HH * 4);
  float* doc_vec  = (float*)take((size_t)BB * 512 * 4);
  float* topic_mat = (float*)take((size_t)BB * TT * 512 * 4);
  float* topic_rep = (float*)take((size_t)M * 512 * 4);
  float* scoresD  = (float*)take((size_t)M * 4);
  float* scoresT  = (float*)take((size_t)M * 4);
  float* wD       = (float*)take((size_t)M * 4);
  float* wT       = (float*)take((size_t)M * 4);
  // gi region (50 MB) is dead after the GRU loop; reuse for cat/inp (48 MB)
  __bf16* catDS = (__bf16*)giF;
  __bf16* catTS = (__bf16*)((char*)giF + (size_t)M * FH * 2);
  __bf16* inp   = (__bf16*)((char*)giF + 2 * (size_t)M * FH * 2);

  hipMemsetAsync(hF, 0, (size_t)BB * HH * 4, stream);
  hipMemsetAsync(hB, 0, (size_t)BB * HH * 4, stream);

  // ---- weight conversion to bf16 [N][Kp] ----------------------------------
  auto cdiv = [](int a, int b) { return (a + b - 1) / b; };
  k_conv_nk<<<cdiv(GG * EP, 256), 256, 0, stream>>>(Wih_f, WihFb, GG, EE, EP);
  k_conv_nk<<<cdiv(GG * EP, 256), 256, 0, stream>>>(Wih_b, WihBb, GG, EE, EP);
  k_conv_nk<<<cdiv(GG * HH, 256), 256, 0, stream>>>(Whh_f, WhhFb, GG, HH, HH);
  k_conv_nk<<<cdiv(GG * HH, 256), 256, 0, stream>>>(Whh_b, WhhBb, GG, HH, HH);
  k_conv_kn_T<<<cdiv(FH * FH, 256), 256, 0, stream>>>(W_att, WattTb, FH, FH, FH);
  k_conv_kn_T<<<cdiv(DD * FH, 256), 256, 0, stream>>>(W_dna, WdnaTb, DD, FH, FH);

  // ---- stage 1: sentence vectors ------------------------------------------
  k_embed_mean<<<M, EP, 0, stream>>>(word_ids, emb, sentA);

  // ---- stage 2: batched GRU input projections (WMMA GEMM) -----------------
  k_gemm_bias<<<dim3(M / 16, GG / 16), 32, 0, stream>>>(sentA, WihFb, bih_f, giF, GG, EP);
  k_gemm_bias<<<dim3(M / 16, GG / 16), 32, 0, stream>>>(sentA, WihBb, bih_b, giB, GG, EP);

  // ---- stage 3: sequential recurrence (both directions per launch) --------
  for (int t = 0; t < SS; ++t)
    k_gru_step<<<dim3(2, 2), 512, 0, stream>>>(t, giF, giB, hF, hB,
                                               WhhFb, WhhBb, bhh_f, bhh_b, sent_rep);

  // ---- stage 4: doc vec, topic rep, concat inputs -------------------------
  k_doc_vec<<<32, 512, 0, stream>>>(hF, hB, doc_vec);
  k_topic_mat<<<BB, 512, 0, stream>>>(sent_rep, tse, topic_mat);
  k_cat<<<M, 512, 0, stream>>>(sent_rep, doc_vec, topic_mat, tse, catDS, catTS, topic_rep);

  // ---- stage 5: fused attention scores + softmax --------------------------
  k_att_score<<<dim3(M / 16, 2), 256, 0, stream>>>(catDS, catTS, WattTb, v_att,
                                                   scoresD, scoresT);
  k_softmax<<<dim3(BB, 2), SS, 0, stream>>>(scoresD, scoresT, wD, wT);

  // ---- stage 6: context, fused dense head -> logits -----------------------
  k_context_inp<<<M, 512, 0, stream>>>(sent_rep, doc_vec, topic_rep, wD, wT, inp);
  k_dense_out<<<M / 16, 256, 0, stream>>>(inp, WdnaTb, b_dna, W_out, b_out, (float*)d_out);
}
